// InertialEncoder_41618233098288
// MI455X (gfx1250) — matrix operand order repaired and checked by reading, not compiled, not run on hardware
//
#include <hip/hip_runtime.h>
#include <cstdint>

// Problem constants (must match the reference).
#define BB  2048
#define TT  512
#define NIN 6
#define HH  32
#define NOUT 64

typedef __attribute__((ext_vector_type(16))) _Float16 v16h;
typedef __attribute__((ext_vector_type(8)))  float    v8f;

union V16HU { v16h v; uint4 q[2]; };

// WMMA D=A*B+C, 16x16x32 f16->f32.
__device__ __forceinline__ v8f wmma32(v16h a, v16h b, v8f c) {
    return __builtin_amdgcn_wmma_f32_16x16x32_f16(
        /*neg_a=*/false, a, /*neg_b=*/false, b,
        /*c_mod=*/(short)0, c, /*reuse_a=*/false, /*reuse_b=*/false);
}

// Native transcendental activations: no IEEE division expansion.
#if __has_builtin(__builtin_amdgcn_tanhf)
__device__ __forceinline__ float tanhx(float x) { return __builtin_amdgcn_tanhf(x); }
#else
__device__ __forceinline__ float tanhx(float x) {
    // tanh(x) = 1 - 2/(exp2(2x*log2e)+1), native v_exp_f32 + v_rcp_f32
    float e = __builtin_amdgcn_exp2f(x * 2.8853900817779268f);
    return 1.0f - 2.0f * __builtin_amdgcn_rcpf(e + 1.0f);
}
#endif
__device__ __forceinline__ float sigm(float x) {
    // sigmoid(x) = 0.5*tanh(x/2) + 0.5  -> one trans + one fma
    return __builtin_fmaf(0.5f, tanhx(0.5f * x), 0.5f);
}
__device__ __forceinline__ float fast_exp(float x) {
    return __builtin_amdgcn_exp2f(x * 1.4426950408889634f);   // v_exp_f32
}

// Build a 32x16 f16 B-tile fragment from a row-major (rows>=16*tl+16, 32) f32 weight
// matrix W, for column-tile tl of W^T (i.e. B[k][n] = W[16*tl+n][k]).
// CDNA5 B-layout (CDNA5 ISA): lane l holds column n = l&15; half j holds
// k = (j>>3)*16 + (l>>4)*8 + (j&7).
__device__ __forceinline__ v16h load_btile_k32(const float* __restrict__ w,
                                               int tl, int nl, int hi) {
    const float* p = w + (size_t)(16 * tl + nl) * 32 + hi * 8;
    v16h b;
#pragma unroll
    for (int j = 0; j < 16; ++j) b[j] = (_Float16)p[j < 8 ? j : j + 8];
    return b;
}

// Fully fused 2-layer LSTM + heads. Block = 64 threads = 2 waves.
// wave0: layer0; wave1: layer1 one step behind, h0_t handed over via a
// double-buffered LDS tile; wave1's epilogue computes both output heads.
//
// A-fragment layout (ISA 16-bit A-matrix 16x32): lane l holds row m=l&15,
// half j -> k = (j>>3)*16 + (l>>4)*8 + (j&7). From a row-major 16x32 f16 tile
// in LDS, lane l's A-fragment is two contiguous 16B loads at
// halves [m*32 + (l>>4)*8] and [m*32 + 16 + (l>>4)*8].
//
// D-fragment layout (f32 16x16): lane l, vgpr r -> (m = r + 8*(l>>4), n = l&15).
__global__ __launch_bounds__(64) void lstm2_fused_kernel(
    const float* __restrict__ imu,
    const float* __restrict__ w_ih0, const float* __restrict__ w_hh0,
    const float* __restrict__ b_ih0, const float* __restrict__ b_hh0,
    const float* __restrict__ w_ih1, const float* __restrict__ w_hh1,
    const float* __restrict__ b_ih1, const float* __restrict__ b_hh1,
    const float* __restrict__ wf, const float* __restrict__ bf,
    const float* __restrict__ wn, const float* __restrict__ bn,
    float* __restrict__ out)
{
    __shared__ __align__(16) _Float16 comm[2][16 * 32];  // layer0 -> layer1 handoff (and wave0 D->A conv)
    __shared__ __align__(16) _Float16 conv1[16 * 32];    // wave1 private D->A conversion buffer

    const int lane = threadIdx.x & 31;
    const int wave = threadIdx.x >> 5;
    const int nl   = lane & 15;
    const int hi   = lane >> 4;
    const int b0   = blockIdx.x * 16;

    const float* wih = wave ? w_ih1 : w_ih0;
    const float* whh = wave ? w_hh1 : w_hh0;
    const float* bih = wave ? b_ih1 : b_ih0;
    const float* bhh = wave ? b_hh1 : b_hh0;

    // Pin weight B-tiles (f16) and pre-splat fused-bias C fragments in VGPRs.
    v16h Bih[8], Bhh[8];
    v8f  bias8[8];
#pragma unroll
    for (int tl = 0; tl < 8; ++tl) {
        Bhh[tl] = load_btile_k32(whh, tl, nl, hi);
        if (wave) {
            Bih[tl] = load_btile_k32(wih, tl, nl, hi);   // (128,32) full K
        } else {
            // w_ih0 is (128,6): k = hi*16 + j valid only for hi==0, j<6; rest zero-pad.
            v16h b;
#pragma unroll
            for (int j = 0; j < 16; ++j) b[j] = (_Float16)0.0f;
            if (hi == 0) {
                const float* p = wih + (size_t)(16 * tl + nl) * NIN;
#pragma unroll
                for (int j = 0; j < NIN; ++j) b[j] = (_Float16)p[j];
            }
            Bih[tl] = b;
        }
        const float bs = bih[16 * tl + nl] + bhh[16 * tl + nl];
#pragma unroll
        for (int r = 0; r < 8; ++r) bias8[tl][r] = bs;
    }

    // Cell state in D-layout (2 column-fragments x 8 rows), hidden in A-layout.
    float c[2][8];
#pragma unroll
    for (int f = 0; f < 2; ++f)
#pragma unroll
        for (int r = 0; r < 8; ++r) c[f][r] = 0.0f;
    v16h hA;
#pragma unroll
    for (int j = 0; j < 16; ++j) hA[j] = (_Float16)0.0f;

    // Software-pipelined imu input: xn holds step t's values, prefetched one
    // step ahead so the global-load latency hides under the activations+barrier.
    float xn[NIN];
    if (wave == 0 && hi == 0) {
        const float* xp = imu + (size_t)(b0 + nl) * TT * NIN;   // t = 0
#pragma unroll
        for (int k = 0; k < NIN; ++k) xn[k] = xp[k];
    }

    for (int it = 0; it <= TT; ++it) {
        __syncthreads();   // protects comm slot handoff between the two waves
        const bool active = wave ? (it >= 1) : (it < TT);
        if (active) {
            const int t = wave ? (it - 1) : it;

            // ---- input A-tile ----
            v16h xA;
            if (wave == 0) {
#pragma unroll
                for (int j = 0; j < 16; ++j) xA[j] = (_Float16)0.0f;
                if (hi == 0) {  // only k=0..5 nonzero -> lanes 0-15, halves 0-5
#pragma unroll
                    for (int k = 0; k < NIN; ++k) xA[k] = (_Float16)xn[k];
                }
                // Prefetch next step's input now; consumed next iteration.
                if (t + 1 < TT && hi == 0) {
                    const float* xp = imu + ((size_t)(b0 + nl) * TT + (t + 1)) * NIN;
#pragma unroll
                    for (int k = 0; k < NIN; ++k) xn[k] = xp[k];
                }
            } else {
                const _Float16* src = comm[t & 1];  // written by wave0 at iteration it-1
                V16HU u;
                u.q[0] = *(const uint4*)(src + nl * 32 + hi * 8);
                u.q[1] = *(const uint4*)(src + nl * 32 + 16 + hi * 8);
                xA = u.v;
            }

            // ---- gates = bias + x@Wih^T + h@Whh^T : 16 WMMAs, C fed from bias8 ----
            v8f acc[8];
#pragma unroll
            for (int tl = 0; tl < 8; ++tl) {
                v8f a = wmma32(xA, Bih[tl], bias8[tl]);
                a     = wmma32(hA, Bhh[tl], a);
                acc[tl] = a;
            }

            // ---- activations + state update; emit h as f16 row-major into LDS ----
            _Float16* dst = wave ? conv1 : comm[it & 1];
#pragma unroll
            for (int f2 = 0; f2 < 2; ++f2) {
#pragma unroll
                for (int r = 0; r < 8; ++r) {
                    float iv = sigm(acc[0 + f2][r]);
                    float fv = sigm(acc[2 + f2][r]);
                    float gv = tanhx(acc[4 + f2][r]);
                    float ov = sigm(acc[6 + f2][r]);
                    float cn = __builtin_fmaf(fv, c[f2][r], iv * gv);
                    c[f2][r] = cn;
                    float hv = ov * tanhx(cn);
                    dst[(r + 8 * hi) * 32 + 16 * f2 + nl] = (_Float16)hv;
                }
            }

            // ---- read back h in A-layout for the next step's recurrence ----
            {
                V16HU u;
                u.q[0] = *(const uint4*)(dst + nl * 32 + hi * 8);
                u.q[1] = *(const uint4*)(dst + nl * 32 + 16 + hi * 8);
                hA = u.v;
            }
        }
    }

    // ---- heads on wave1's final hidden state (hA == h1_{T-1} in A-layout) ----
    // Wave-uniform branch: EXEC is all ones for wave1, so WMMA is legal here.
    if (wave == 1) {
        float* outA = out;
        float* outR = out + (size_t)BB * NOUT;
#pragma unroll
        for (int tl = 0; tl < 4; ++tl) {
            v16h Bf = load_btile_k32(wf, tl, nl, hi);
            v16h Bn = load_btile_k32(wn, tl, nl, hi);
            const float ba = bf[16 * tl + nl];
            const float br = bn[16 * tl + nl];
            v8f a, rr;
#pragma unroll
            for (int r = 0; r < 8; ++r) { a[r] = ba; rr[r] = br; }
            a  = wmma32(hA, Bf, a);
            rr = wmma32(hA, Bn, rr);
#pragma unroll
            for (int r = 0; r < 8; ++r) {
                const int row = b0 + r + 8 * hi;
                outA[(size_t)row * NOUT + 16 * tl + nl] = a[r];
                outR[(size_t)row * NOUT + 16 * tl + nl] = fast_exp(rr[r]);
            }
        }
    }
}

extern "C" void kernel_launch(void* const* d_in, const int* in_sizes, int n_in,
                              void* d_out, int out_size, void* d_ws, size_t ws_size,
                              hipStream_t stream) {
    const float* imu   = (const float*)d_in[0];
    const float* w_ih0 = (const float*)d_in[1];
    const float* w_hh0 = (const float*)d_in[2];
    const float* b_ih0 = (const float*)d_in[3];
    const float* b_hh0 = (const float*)d_in[4];
    const float* w_ih1 = (const float*)d_in[5];
    const float* w_hh1 = (const float*)d_in[6];
    const float* b_ih1 = (const float*)d_in[7];
    const float* b_hh1 = (const float*)d_in[8];
    const float* wf    = (const float*)d_in[9];
    const float* bf    = (const float*)d_in[10];
    const float* wn    = (const float*)d_in[11];
    const float* bn    = (const float*)d_in[12];

    (void)d_ws; (void)ws_size; (void)in_sizes; (void)n_in; (void)out_size;

    const int ntiles = BB / 16;          // 128 workgroups, 2 waves each
    lstm2_fused_kernel<<<ntiles, 64, 0, stream>>>(
        imu, w_ih0, w_hh0, b_ih0, b_hh0, w_ih1, w_hh1, b_ih1, b_hh1,
        wf, bf, wn, bn, (float*)d_out);
}
